// AttentionBlock_64063732187703
// MI455X (gfx1250) — matrix operand order repaired and checked by reading, not compiled, not run on hardware
//
#include <hip/hip_runtime.h>
#include <hip/hip_bf16.h>
#include <math.h>

typedef __attribute__((ext_vector_type(16))) __bf16 v16bf;
typedef __attribute__((ext_vector_type(8)))  __bf16 v8bf;
typedef __attribute__((ext_vector_type(4)))  __bf16 v4bf;
typedef __attribute__((ext_vector_type(8)))  float  v8f;

#define NB 4
#define NT 4096
#define ND 512
#define NN 512           // K == V == 512
#define MASKVAL (-3.0e4f)
#define SQRTK 22.62741699796952f

// ---------- WMMA fragment helpers (16x16x32 bf16, contiguous k order shared by A and B) ----------
// A operand: 16(M) x 32(K), row-major source [M, K] with leading dim ld.
// lane m = l%16 holds row m; lanes<16: K = 0..15 ; lanes>=16: K = 16..31.
__device__ __forceinline__ v16bf frag_a(const __bf16* p0, int ld, int lane) {
  const __bf16* p = p0 + (size_t)(lane & 15) * ld + ((lane >> 4) << 4);
  union { v16bf v; v8bf h[2]; } u;
  u.h[0] = *(const v8bf*)(p);
  u.h[1] = *(const v8bf*)(p + 8);
  return u.v;
}
// B operand: 32(K) x 16(N), source stored TRANSPOSED [N, K] with leading dim ld.
// lane holds column n = l%16 ; lanes<16: K = 0..15 ; lanes>=16: K = 16..31.
__device__ __forceinline__ v16bf frag_b(const __bf16* p0, int ld, int lane) {
  const __bf16* p = p0 + (size_t)(lane & 15) * ld + ((lane >> 4) << 4);
  union { v16bf v; v8bf h[2]; } u;
  u.h[0] = *(const v8bf*)(p);
  u.h[1] = *(const v8bf*)(p + 8);
  return u.v;
}
__device__ __forceinline__ v8f wmma32(v16bf a, v16bf b, v8f c) {
  return __builtin_amdgcn_wmma_f32_16x16x32_bf16(false, a, false, b, (short)0, c, false, false);
}

// ---------- kernel 0a: x fp32 -> bf16 ----------
__global__ void k_cvt_x(const float* __restrict__ x, __bf16* __restrict__ xb) {
  int i = (blockIdx.x * blockDim.x + threadIdx.x) * 4;
  float4 f = *(const float4*)(x + i);
  v4bf o;
  o.x = (__bf16)f.x; o.y = (__bf16)f.y; o.z = (__bf16)f.z; o.w = (__bf16)f.w;
  *(v4bf*)(xb + i) = o;
}

// ---------- kernel 0b: W [D,N] fp32 -> Wt [N,D] bf16, for k/q/v ----------
__global__ void k_cvt_w(const float* __restrict__ Wk, const float* __restrict__ Wq,
                        const float* __restrict__ Wv, __bf16* __restrict__ Wt) {
  int idx = blockIdx.x * blockDim.x + threadIdx.x;     // 3*512*512 total
  int w = idx >> 18;
  int r = idx & 262143;
  int n = r >> 9, d = r & 511;
  const float* src = (w == 0) ? Wk : ((w == 1) ? Wq : Wv);
  Wt[idx] = (__bf16)src[d * ND + n];                   // Wt[w][n][d]
}

// ---------- kernel 1: QKV projection GEMM (x @ W + b), bf16 WMMA ----------
// z=0 -> K (row-major [b*T, N]); z=1 -> Q (row-major); z=2 -> V stored transposed [b][v][t]
// Wave tile: 32M x 64N (8 acc tiles). Per K-step: 2 A + 4 B fragments feed 8 WMMAs.
__global__ __launch_bounds__(256) void k_qkv(const __bf16* __restrict__ xb,
                                             const __bf16* __restrict__ Wt,
                                             const float* __restrict__ bk,
                                             const float* __restrict__ bq,
                                             const float* __restrict__ bv,
                                             __bf16* __restrict__ Qb,
                                             __bf16* __restrict__ Kb,
                                             __bf16* __restrict__ Vt) {
  int lane = threadIdx.x & 31, wave = threadIdx.x >> 5;
  int z = blockIdx.z;
  int tileM = blockIdx.x * 256 + wave * 32;
  int tileN = blockIdx.y * 64;
  const __bf16* W = Wt + (size_t)z * (ND * NN);
  const float* bias = (z == 0) ? bk : ((z == 1) ? bq : bv);

  v8f acc[8];                                      // [mt][i] = acc[mt*4+i]
#pragma unroll
  for (int i = 0; i < 4; i++) {
    float bv_ = bias[tileN + i * 16 + (lane & 15)];
#pragma unroll
    for (int j = 0; j < 8; j++) { acc[i][j] = bv_; acc[4 + i][j] = bv_; }
  }
  const __bf16* arow0 = xb + (size_t)tileM * ND;
  const __bf16* arow1 = arow0 + (size_t)16 * ND;

  // prologue: fragments for k0 = 0
  v16bf fa0 = frag_a(arow0, ND, lane);
  v16bf fa1 = frag_a(arow1, ND, lane);
  v16bf fb[4];
#pragma unroll
  for (int i = 0; i < 4; i++)
    fb[i] = frag_b(W + (size_t)(tileN + i * 16) * ND, ND, lane);

#pragma unroll
  for (int k0 = 0; k0 < ND - 32; k0 += 32) {
    // prefetch next K step while current WMMAs execute
    v16bf na0 = frag_a(arow0 + k0 + 32, ND, lane);
    v16bf na1 = frag_a(arow1 + k0 + 32, ND, lane);
    v16bf nb[4];
#pragma unroll
    for (int i = 0; i < 4; i++)
      nb[i] = frag_b(W + (size_t)(tileN + i * 16) * ND + k0 + 32, ND, lane);
#pragma unroll
    for (int i = 0; i < 4; i++) {
      acc[i]     = wmma32(fa0, fb[i], acc[i]);
      acc[4 + i] = wmma32(fa1, fb[i], acc[4 + i]);
    }
    fa0 = na0; fa1 = na1;
#pragma unroll
    for (int i = 0; i < 4; i++) fb[i] = nb[i];
  }
#pragma unroll
  for (int i = 0; i < 4; i++) {
    acc[i]     = wmma32(fa0, fb[i], acc[i]);
    acc[4 + i] = wmma32(fa1, fb[i], acc[4 + i]);
  }

  if (z < 2) {
    __bf16* out = (z == 0) ? Kb : Qb;
#pragma unroll
    for (int mt = 0; mt < 2; mt++) {
      int mrow = tileM + mt * 16 + ((lane >> 4) << 3);
#pragma unroll
      for (int i = 0; i < 4; i++)
#pragma unroll
        for (int j = 0; j < 8; j++)
          out[(size_t)(mrow + j) * NN + tileN + i * 16 + (lane & 15)] =
              (__bf16)acc[mt * 4 + i][j];
    }
  } else {
    int bb = tileM >> 12;                          // batch (256-row blocks stay in batch)
#pragma unroll
    for (int mt = 0; mt < 2; mt++) {
      int tl = (tileM & (NT - 1)) + mt * 16 + ((lane >> 4) << 3);
#pragma unroll
      for (int i = 0; i < 4; i++) {
        v8bf pk;
#pragma unroll
        for (int j = 0; j < 8; j++) pk[j] = (__bf16)acc[mt * 4 + i][j];
        int vcol = tileN + i * 16 + (lane & 15);
        *(v8bf*)(Vt + ((size_t)bb * NN + vcol) * NT + tl) = pk;  // contiguous t
      }
    }
  }
}

// ---------- kernel 2: column softmax stats (softmax over query axis t) ----------
// For each column s: colmax = max_{t>=s} logit, colsum = sum exp(.-colmax).
// Stores colmax and 1/(colsum*sqrt(K)). Each wave handles 32 t-rows per step
// (two independent WMMA chains sharing one B fragment) with load prefetch.
__global__ __launch_bounds__(256) void k_colstats(const __bf16* __restrict__ Qb,
                                                  const __bf16* __restrict__ Kb,
                                                  float* __restrict__ cmax,
                                                  float* __restrict__ cinv) {
  int lane = threadIdx.x & 31, wave = threadIdx.x >> 5;
  int sb = blockIdx.x * 64;
  int b  = blockIdx.y;
  int ss = wave & 3, ts = wave >> 2;
  int scol = sb + ss * 16;
  const __bf16* Qbase = Qb + (size_t)b * NT * NN;
  const __bf16* krow  = Kb + (size_t)b * NT * NN + (size_t)scol * NN;
  float rm = MASKVAL, rs = 0.0f;
  int n = scol + (lane & 15);

  for (int t0 = sb; t0 < NT; t0 += 64) {
    int trow = t0 + ts * 32;                       // this wave: rows trow..trow+31
    const __bf16* q0 = Qbase + (size_t)trow * NN;
    const __bf16* q1 = q0 + (size_t)16 * NN;

    v8f ac0, ac1;
#pragma unroll
    for (int j = 0; j < 8; j++) { ac0[j] = 0.0f; ac1[j] = 0.0f; }
    v16bf fa0 = frag_a(q0, NN, lane);
    v16bf fa1 = frag_a(q1, NN, lane);
    v16bf fb  = frag_b(krow, NN, lane);
#pragma unroll
    for (int k0 = 0; k0 < NN - 32; k0 += 32) {
      v16bf na0 = frag_a(q0 + k0 + 32, NN, lane);
      v16bf na1 = frag_a(q1 + k0 + 32, NN, lane);
      v16bf nb  = frag_b(krow + k0 + 32, NN, lane);
      ac0 = wmma32(fa0, fb, ac0);
      ac1 = wmma32(fa1, fb, ac1);
      fa0 = na0; fa1 = na1; fb = nb;
    }
    ac0 = wmma32(fa0, fb, ac0);
    ac1 = wmma32(fa1, fb, ac1);

    int m0 = trow + ((lane >> 4) << 3);
    float lv[16], tmax = MASKVAL;
#pragma unroll
    for (int j = 0; j < 8; j++) {
      float l = ac0[j];
      if (m0 + j < n) l = MASKVAL;                 // causal: only t >= s valid
      lv[j] = l;
      tmax = fmaxf(tmax, l);
    }
#pragma unroll
    for (int j = 0; j < 8; j++) {
      float l = ac1[j];
      if (m0 + 16 + j < n) l = MASKVAL;
      lv[8 + j] = l;
      tmax = fmaxf(tmax, l);
    }
    float tsum = 0.0f;
#pragma unroll
    for (int j = 0; j < 16; j++) tsum += __expf(lv[j] - tmax);
    // combine the two 16-lane halves (rows m and m+8)
    float om = __shfl_xor(tmax, 16, 32);
    float os = __shfl_xor(tsum, 16, 32);
    float cmx = fmaxf(tmax, om);
    float csm = tsum * __expf(tmax - cmx) + os * __expf(om - cmx);
    // online update of running (max, sum)
    float nm = fmaxf(rm, cmx);
    rs = rs * __expf(rm - nm) + csm * __expf(cmx - nm);
    rm = nm;
  }
  __shared__ float smax[8][16], ssum[8][16];
  if (lane < 16) { smax[wave][lane] = rm; ssum[wave][lane] = rs; }
  __syncthreads();
  if (wave < 4 && lane < 16) {                     // merge ts=0 and ts=1 partners
    float m0v = smax[wave][lane],     s0v = ssum[wave][lane];
    float m1v = smax[wave + 4][lane], s1v = ssum[wave + 4][lane];
    float nm = fmaxf(m0v, m1v);
    float s  = s0v * __expf(m0v - nm) + s1v * __expf(m1v - nm);
    int col = sb + wave * 16 + lane;
    cmax[(size_t)b * NT + col] = nm;
    cinv[(size_t)b * NT + col] = 1.0f / (s * SQRTK);
  }
}

// ---------- kernel 3: P = exp(logit-cmax)*cinv ; out = x + P @ V ----------
// 64-row t-block per workgroup. 8 waves: wave w computes logit subtile
// (tt=w&3, sh=w>>2) -> bf16 P tile in LDS; then each wave does PV for its own
// t-subtile over v-groups {sh + 2i}. No redundant logit recompute.
__global__ __launch_bounds__(256) void k_pv(const __bf16* __restrict__ Qb,
                                            const __bf16* __restrict__ Kb,
                                            const __bf16* __restrict__ Vt,
                                            const float* __restrict__ cmax,
                                            const float* __restrict__ cinv,
                                            const float* __restrict__ x,
                                            float* __restrict__ out) {
  int lane = threadIdx.x & 31, wave = threadIdx.x >> 5;
  int tb = blockIdx.x * 64;
  int b  = blockIdx.y;
  int tt = wave & 3;
  int sh = wave >> 2;
  const __bf16* Qbase = Qb + (size_t)b * NT * NN;
  const __bf16* Kbase = Kb + (size_t)b * NT * NN;
  const __bf16* Vbase = Vt + (size_t)b * NN * NT;   // [v][t]
  __shared__ __bf16 P[64][32];

  v8f acc[16];
#pragma unroll
  for (int i = 0; i < 16; i++)
#pragma unroll
    for (int j = 0; j < 8; j++) acc[i][j] = 0.0f;

  int trow = tb + tt * 16;
  int mloc = ((lane >> 4) << 3);
  const __bf16* qrow = Qbase + (size_t)trow * NN;

  for (int s0 = 0; s0 < tb + 64; s0 += 32) {
    // --- step 1: this wave's 16x16 logit subtile (pipelined loads) ---
    int scol = s0 + sh * 16;
    int sgl = scol + (lane & 15);
    float cm = cmax[(size_t)b * NT + sgl];          // issued early, used post-loop
    float iv = cinv[(size_t)b * NT + sgl];
    const __bf16* krow = Kbase + (size_t)scol * NN;

    v8f lacc;
#pragma unroll
    for (int j = 0; j < 8; j++) lacc[j] = 0.0f;
    v16bf fa = frag_a(qrow, NN, lane), fb = frag_b(krow, NN, lane);
#pragma unroll
    for (int k0 = 0; k0 < NN - 32; k0 += 32) {
      v16bf na = frag_a(qrow + k0 + 32, NN, lane);
      v16bf nb = frag_b(krow + k0 + 32, NN, lane);
      lacc = wmma32(fa, fb, lacc);
      fa = na; fb = nb;
    }
    lacc = wmma32(fa, fb, lacc);

#pragma unroll
    for (int j = 0; j < 8; j++) {
      int t = trow + mloc + j;
      float p = (sgl <= t) ? __expf(lacc[j] - cm) * iv : 0.0f;
      P[tt * 16 + mloc + j][sh * 16 + (lane & 15)] = (__bf16)p;
    }
    __syncthreads();
    // --- step 2: PV with P as A-operand from LDS, Vt as B-operand (pipelined) ---
    v16bf a = frag_a(&P[tt * 16][0], 32, lane);
    v16bf bcur = frag_b(Vbase + (size_t)(sh * 64) * NT + s0, NT, lane);
#pragma unroll
    for (int u = 0; u < 16; u++) {
      v16bf bnext = bcur;
      if (u < 15) {
        int un = u + 1;
        int vcoln = (sh + 2 * (un >> 2)) * 64 + (un & 3) * 16;
        bnext = frag_b(Vbase + (size_t)vcoln * NT + s0, NT, lane);
      }
      acc[u] = wmma32(a, bcur, acc[u]);
      bcur = bnext;
    }
    __syncthreads();
  }
  // --- epilogue: out = x + read ---
#pragma unroll
  for (int i = 0; i < 4; i++) {
    int vg = sh + 2 * i;
#pragma unroll
    for (int j2 = 0; j2 < 4; j2++) {
      int vcol = vg * 64 + j2 * 16 + (lane & 15);
#pragma unroll
      for (int j = 0; j < 8; j++) {
        int t = trow + mloc + j;
        size_t idx = ((size_t)b * NT + t) * NN + vcol;
        out[idx] = x[idx] + acc[i * 4 + j2][j];
      }
    }
  }
}

// ---------- host-side launch ----------
extern "C" void kernel_launch(void* const* d_in, const int* in_sizes, int n_in,
                              void* d_out, int out_size, void* d_ws, size_t ws_size,
                              hipStream_t stream) {
  const float* x  = (const float*)d_in[0];
  const float* Wk = (const float*)d_in[1];
  const float* bk = (const float*)d_in[2];
  const float* Wq = (const float*)d_in[3];
  const float* bq = (const float*)d_in[4];
  const float* Wv = (const float*)d_in[5];
  const float* bv = (const float*)d_in[6];
  float* out = (float*)d_out;

  char* w = (char*)d_ws;
  __bf16* xb = (__bf16*)(w);                        // 16 MB
  __bf16* Wt = (__bf16*)(w + 16777216);             // 1.5 MB (3x[N,D] bf16)
  __bf16* Qb = (__bf16*)(w + 18350080);             // 16 MB  [b][t][k]
  __bf16* Kb = (__bf16*)(w + 35127296);             // 16 MB  [b][s][k]
  __bf16* Vt = (__bf16*)(w + 51904512);             // 16 MB  [b][v][t]
  float*  cm = (float*)(w + 68681728);              // 64 KB
  float*  ci = (float*)(w + 68747264);              // 64 KB

  k_cvt_x<<<dim3((NB * NT * ND) / (256 * 4)), dim3(256), 0, stream>>>(x, xb);
  k_cvt_w<<<dim3((3 * ND * NN) / 256), dim3(256), 0, stream>>>(Wk, Wq, Wv, Wt);
  k_qkv<<<dim3((NB * NT) / 256, NN / 64, 3), dim3(256), 0, stream>>>(
      xb, Wt, bk, bq, bv, Qb, Kb, Vt);
  k_colstats<<<dim3(NT / 64, NB), dim3(256), 0, stream>>>(Qb, Kb, cm, ci);
  k_pv<<<dim3(NT / 64, NB), dim3(256), 0, stream>>>(Qb, Kb, Vt, cm, ci, x, out);
}
